// tsh_mid_block_58420145160921
// MI455X (gfx1250) — compile-verified, matrix operand order checked by reading, not run on hardware
//
#include <hip/hip_runtime.h>
#include <stdint.h>

// ---------------- types ----------------
typedef __bf16 bf16_t;
typedef __attribute__((ext_vector_type(16))) __bf16 v16bf;
typedef __attribute__((ext_vector_type(8)))  float   v8f;

__device__ __forceinline__ bf16_t f2bf(float f) {
    union { float f; uint32_t u; } v; v.f = f;
    uint32_t r = v.u + 0x7FFFu + ((v.u >> 16) & 1u);   // round-to-nearest-even
    union { uint16_t u; bf16_t b; } o; o.u = (uint16_t)(r >> 16);
    return o.b;
}

// ---------------- fp32 -> bf16 convert ----------------
__global__ __launch_bounds__(256) void k_f32_to_bf16(const float* __restrict__ in,
                                                     bf16_t* __restrict__ out, long n) {
    long i = (long)blockIdx.x * blockDim.x + threadIdx.x;
    long stride = (long)gridDim.x * blockDim.x;
    for (; i < n; i += stride) out[i] = f2bf(in[i]);
}

// ---------------- generic WMMA GEMM ----------------
// out[M,N] = act( A[M,K] @ op(B) + bias )
//   NT=true : B is [N,K] row-major (out = A @ B^T)   -- weight GEMMs, Q@K^T
//   NT=false: B is [K,N] row-major (out = A @ B)     -- P@V
// Tiles: block 128x128, K-step 32. 8 waves (wave32), each wave -> 64x32 outputs
// via eight v_wmma_f32_16x16x32_bf16.
template <bool NT, bool SILU>
__global__ __launch_bounds__(256)
void k_gemm_wmma(const bf16_t* __restrict__ A, const bf16_t* __restrict__ B,
                 const float* __restrict__ bias,
                 float* __restrict__ outF, bf16_t* __restrict__ outB,
                 int M, int N, int K,
                 long aBatch, long bBatch, long oBatch)
{
    __shared__ bf16_t As[128][40];   // pad to 80B rows: conflict-free b128 LDS reads
    __shared__ bf16_t Bs[128][40];   // stored [n][k] for both variants

    const int bz = blockIdx.z;
    A += (long)bz * aBatch;
    B += (long)bz * bBatch;
    const long obase = (long)bz * oBatch;

    const int bm0 = blockIdx.y * 128;
    const int bn0 = blockIdx.x * 128;
    const int tid  = threadIdx.x;
    const int lane = tid & 31;
    const int wave = tid >> 5;
    const int wm = (wave >> 2) * 64;   // wave's M offset in block tile (0 / 64)
    const int wn = (wave & 3) * 32;    // wave's N offset (0/32/64/96)
    const int hsel = lane >> 4;        // which 16-wide K half this lane holds
    const int l16  = lane & 15;

    v8f acc[4][2];
    const v8f vzero = {0.f,0.f,0.f,0.f,0.f,0.f,0.f,0.f};
#pragma unroll
    for (int i = 0; i < 4; ++i)
#pragma unroll
        for (int j = 0; j < 2; ++j) acc[i][j] = vzero;

    for (int kk = 0; kk < K; kk += 32) {
        // ---- A tile: 128 rows x 32 cols, 32B per thread ----
        {
            const int row = tid >> 1;
            const int col = (tid & 1) * 16;
            const int4* s = reinterpret_cast<const int4*>(A + (long)(bm0 + row) * K + kk + col);
            int4* d = reinterpret_cast<int4*>(&As[row][col]);
            d[0] = s[0]; d[1] = s[1];
        }
        // ---- B tile ----
        if (NT) {
            const int row = tid >> 1;            // n index
            const int col = (tid & 1) * 16;      // k index
            const int4* s = reinterpret_cast<const int4*>(B + (long)(bn0 + row) * K + kk + col);
            int4* d = reinterpret_cast<int4*>(&Bs[row][col]);
            d[0] = s[0]; d[1] = s[1];
        } else {
            // B is [K,N]; transpose into Bs[n][k]
            const int k  = tid & 31;
            const int n0 = (tid >> 5) * 16;
            const bf16_t* s = B + (long)(kk + k) * N + bn0 + n0;
#pragma unroll
            for (int i = 0; i < 16; ++i) Bs[n0 + i][k] = s[i];
        }
        __syncthreads();

        v16bf af[4], bfv[2];
#pragma unroll
        for (int i = 0; i < 4; ++i)
            af[i] = *reinterpret_cast<const v16bf*>(&As[wm + i * 16 + l16][hsel * 16]);
#pragma unroll
        for (int j = 0; j < 2; ++j)
            bfv[j] = *reinterpret_cast<const v16bf*>(&Bs[wn + j * 16 + l16][hsel * 16]);

#pragma unroll
        for (int i = 0; i < 4; ++i)
#pragma unroll
            for (int j = 0; j < 2; ++j)
                acc[i][j] = __builtin_amdgcn_wmma_f32_16x16x32_bf16(
                    false, af[i], false, bfv[j], (short)0, acc[i][j], false, false);

        __syncthreads();
    }

    // ---- epilogue: 16x16 f32 tile layout: VGPR r, lanes0-15 -> M=r, lanes16-31 -> M=r+8 ----
#pragma unroll
    for (int i = 0; i < 4; ++i) {
#pragma unroll
        for (int j = 0; j < 2; ++j) {
            const int ncol = bn0 + wn + j * 16 + l16;
            const float bv = bias ? bias[ncol] : 0.0f;
#pragma unroll
            for (int r = 0; r < 8; ++r) {
                const int mrow = bm0 + wm + i * 16 + r + hsel * 8;
                float v = acc[i][j][r] + bv;
                if (SILU) v = v / (1.0f + __expf(-v));
                const long idx = obase + (long)mrow * N + ncol;
                if (outF) outF[idx] = v;
                if (outB) outB[idx] = f2bf(v);
            }
        }
    }
}

// ---------------- L2 normalize rows -> bf16 (one wave per row) ----------------
__global__ __launch_bounds__(256)
void k_l2norm_bf16(const float* __restrict__ in, bf16_t* __restrict__ out, int D) {
    const long row  = (long)blockIdx.x * 8 + (threadIdx.x >> 5);
    const int  lane = threadIdx.x & 31;
    const float* r = in + row * D;
    float ss = 0.0f;
    for (int c = lane; c < D; c += 32) { float v = r[c]; ss += v * v; }
#pragma unroll
    for (int off = 16; off; off >>= 1) ss += __shfl_xor(ss, off, 32);
    const float scale = 1.0f / fmaxf(sqrtf(ss), 1e-12f);
    bf16_t* o = out + row * D;
    for (int c = lane; c < D; c += 32) o[c] = f2bf(r[c] * scale);
}

// ---------------- row softmax -> bf16 (one block per row) ----------------
__global__ __launch_bounds__(256)
void k_softmax_bf16(const float* __restrict__ in, bf16_t* __restrict__ out, int N) {
    __shared__ float sm[8], ss[8];
    const long row = blockIdx.x;
    const float* r = in + row * N;
    const int tid = threadIdx.x, lane = tid & 31, wv = tid >> 5;

    float mx = -__builtin_inff();
    for (int c = tid; c < N; c += 256) mx = fmaxf(mx, r[c]);
#pragma unroll
    for (int off = 16; off; off >>= 1) mx = fmaxf(mx, __shfl_xor(mx, off, 32));
    if (lane == 0) sm[wv] = mx;
    __syncthreads();
    float allm = -__builtin_inff();
#pragma unroll
    for (int i = 0; i < 8; ++i) allm = fmaxf(allm, sm[i]);

    float sum = 0.0f;
    for (int c = tid; c < N; c += 256) sum += __expf(r[c] - allm);
#pragma unroll
    for (int off = 16; off; off >>= 1) sum += __shfl_xor(sum, off, 32);
    if (lane == 0) ss[wv] = sum;
    __syncthreads();
    float tot = 0.0f;
#pragma unroll
    for (int i = 0; i < 8; ++i) tot += ss[i];
    const float inv = 1.0f / tot;

    bf16_t* o = out + row * N;
    for (int c = tid; c < N; c += 256) o[c] = f2bf(__expf(r[c] - allm) * inv);
}

// ---------------- residual add + LayerNorm: res=a+b (f32), nrm=LN(res)*g+be (bf16) ----------------
__global__ __launch_bounds__(256)
void k_res_ln(const float* __restrict__ a, const float* __restrict__ b,
              const float* __restrict__ g, const float* __restrict__ be,
              float* __restrict__ res, bf16_t* __restrict__ nrm, int D) {
    __shared__ float s1[8], s2[8];
    const long row = blockIdx.x;
    const float* ra = a + row * D;
    const float* rb = b + row * D;
    float*  rr = res + row * D;
    bf16_t* rn = nrm + row * D;
    const int tid = threadIdx.x, lane = tid & 31, wv = tid >> 5;

    float sum = 0.0f, sq = 0.0f;
    for (int c = tid; c < D; c += 256) {
        float v = ra[c] + rb[c];
        rr[c] = v;
        sum += v; sq += v * v;
    }
#pragma unroll
    for (int off = 16; off; off >>= 1) {
        sum += __shfl_xor(sum, off, 32);
        sq  += __shfl_xor(sq,  off, 32);
    }
    if (lane == 0) { s1[wv] = sum; s2[wv] = sq; }
    __syncthreads();
    float ts = 0.0f, tq = 0.0f;
#pragma unroll
    for (int i = 0; i < 8; ++i) { ts += s1[i]; tq += s2[i]; }
    const float mean = ts / (float)D;
    const float var  = tq / (float)D - mean * mean;
    const float rstd = rsqrtf(var + 1e-5f);
    for (int c = tid; c < D; c += 256)
        rn[c] = f2bf((rr[c] - mean) * rstd * g[c] + be[c]);
}

// ---------------- host orchestration ----------------
extern "C" void kernel_launch(void* const* d_in, const int* in_sizes, int n_in,
                              void* d_out, int out_size, void* d_ws, size_t ws_size,
                              hipStream_t stream) {
    (void)in_sizes; (void)n_in; (void)out_size; (void)ws_size;
    const long Bb = 4, S = 2048, DIN = 256, MID = 1024, FFN = 4096, DOUT = 512;
    const long BS = Bb * S;  // 8192

    const float* x     = (const float*)d_in[0];
    const float* w_emb = (const float*)d_in[1];
    const float* b_emb = (const float*)d_in[2];
    const float* wq    = (const float*)d_in[3];
    const float* bq    = (const float*)d_in[4];
    const float* wk    = (const float*)d_in[5];
    const float* bk    = (const float*)d_in[6];
    const float* wv    = (const float*)d_in[7];
    const float* bv    = (const float*)d_in[8];
    const float* g1    = (const float*)d_in[9];
    const float* be1   = (const float*)d_in[10];
    const float* w_f1  = (const float*)d_in[11];
    const float* b_f1  = (const float*)d_in[12];
    const float* w_f2  = (const float*)d_in[13];
    const float* b_f2  = (const float*)d_in[14];
    const float* g2    = (const float*)d_in[15];
    const float* be2   = (const float*)d_in[16];
    const float* w_h   = (const float*)d_in[17];
    const float* b_h   = (const float*)d_in[18];

    // workspace allocator (256B aligned)
    size_t off = 0;
    auto alloc = [&](size_t bytes) -> char* {
        size_t r = off;
        off = (off + bytes + 255) & ~(size_t)255;
        return (char*)d_ws + r;
    };
    bf16_t* xb      = (bf16_t*)alloc(BS * DIN * 2);
    bf16_t* wemb_b  = (bf16_t*)alloc(MID * DIN * 2);
    bf16_t* wq_b    = (bf16_t*)alloc(MID * MID * 2);
    bf16_t* wk_b    = (bf16_t*)alloc(MID * MID * 2);
    bf16_t* wv_b    = (bf16_t*)alloc(MID * MID * 2);
    bf16_t* wf1_b   = (bf16_t*)alloc(FFN * MID * 2);
    bf16_t* wf2_b   = (bf16_t*)alloc(MID * FFN * 2);
    bf16_t* wh_b    = (bf16_t*)alloc(DOUT * MID * 2);
    float*  embed_f = (float*) alloc(BS * MID * 4);
    bf16_t* embed_b = (bf16_t*)alloc(BS * MID * 2);
    float*  q_f     = (float*) alloc(BS * MID * 4);
    float*  k_f     = (float*) alloc(BS * MID * 4);
    float*  v_f     = (float*) alloc(BS * MID * 4);
    bf16_t* v_b     = (bf16_t*)alloc(BS * MID * 2);
    bf16_t* nq_b    = (bf16_t*)alloc(BS * MID * 2);
    bf16_t* nk_b    = (bf16_t*)alloc(BS * MID * 2);
    float*  sc_f    = (float*) alloc(Bb * S * S * 4);
    bf16_t* pr_b    = (bf16_t*)alloc(Bb * S * S * 2);
    float*  attn_f  = (float*) alloc(BS * MID * 4);
    float*  res1_f  = (float*) alloc(BS * MID * 4);
    bf16_t* nrm1_b  = (bf16_t*)alloc(BS * MID * 2);
    bf16_t* h_b     = (bf16_t*)alloc(BS * FFN * 2);
    float*  ffn_f   = (float*) alloc(BS * MID * 4);
    float*  res2_f  = (float*) alloc(BS * MID * 4);
    bf16_t* nrm2_b  = (bf16_t*)alloc(BS * MID * 2);

    const dim3 blk(256);
    auto cvt = [&](const float* src, bf16_t* dst, long n) {
        int g = (int)((n + 256 * 8 - 1) / (256 * 8));
        k_f32_to_bf16<<<dim3(g), blk, 0, stream>>>(src, dst, n);
    };
    cvt(x, xb, BS * DIN);
    cvt(w_emb, wemb_b, MID * DIN);
    cvt(wq, wq_b, MID * MID);
    cvt(wk, wk_b, MID * MID);
    cvt(wv, wv_b, MID * MID);
    cvt(w_f1, wf1_b, FFN * MID);
    cvt(w_f2, wf2_b, MID * FFN);
    cvt(w_h, wh_b, DOUT * MID);

    // embed: [BS,DIN] @ w_emb[MID,DIN]^T + b_emb
    k_gemm_wmma<true, false><<<dim3(MID / 128, BS / 128, 1), blk, 0, stream>>>(
        xb, wemb_b, b_emb, embed_f, embed_b, (int)BS, (int)MID, (int)DIN, 0, 0, 0);
    // q, k, v projections
    k_gemm_wmma<true, false><<<dim3(MID / 128, BS / 128, 1), blk, 0, stream>>>(
        embed_b, wq_b, bq, q_f, nullptr, (int)BS, (int)MID, (int)MID, 0, 0, 0);
    k_gemm_wmma<true, false><<<dim3(MID / 128, BS / 128, 1), blk, 0, stream>>>(
        embed_b, wk_b, bk, k_f, nullptr, (int)BS, (int)MID, (int)MID, 0, 0, 0);
    k_gemm_wmma<true, false><<<dim3(MID / 128, BS / 128, 1), blk, 0, stream>>>(
        embed_b, wv_b, bv, v_f, v_b, (int)BS, (int)MID, (int)MID, 0, 0, 0);
    // L2 normalize q, k
    k_l2norm_bf16<<<dim3(BS / 8), blk, 0, stream>>>(q_f, nq_b, (int)MID);
    k_l2norm_bf16<<<dim3(BS / 8), blk, 0, stream>>>(k_f, nk_b, (int)MID);
    // scores = nq @ nk^T  (batched, per-batch 2048x2048, K=1024)
    k_gemm_wmma<true, false><<<dim3(S / 128, S / 128, Bb), blk, 0, stream>>>(
        nq_b, nk_b, nullptr, sc_f, nullptr, (int)S, (int)S, (int)MID,
        S * MID, S * MID, S * S);
    // softmax rows
    k_softmax_bf16<<<dim3(Bb * S), blk, 0, stream>>>(sc_f, pr_b, (int)S);
    // attn = probs @ v  (batched NN, K=2048, N=1024)
    k_gemm_wmma<false, false><<<dim3(MID / 128, S / 128, Bb), blk, 0, stream>>>(
        pr_b, v_b, nullptr, attn_f, nullptr, (int)S, (int)MID, (int)S,
        S * S, S * MID, S * MID);
    // res1 = embed + attn ; norm1 = LN(res1)
    k_res_ln<<<dim3(BS), blk, 0, stream>>>(embed_f, attn_f, g1, be1, res1_f, nrm1_b, (int)MID);
    // h = silu(norm1 @ w_f1^T + b_f1)  -> bf16 only
    k_gemm_wmma<true, true><<<dim3(FFN / 128, BS / 128, 1), blk, 0, stream>>>(
        nrm1_b, wf1_b, b_f1, nullptr, h_b, (int)BS, (int)FFN, (int)MID, 0, 0, 0);
    // ffn = h @ w_f2^T + b_f2
    k_gemm_wmma<true, false><<<dim3(MID / 128, BS / 128, 1), blk, 0, stream>>>(
        h_b, wf2_b, b_f2, ffn_f, nullptr, (int)BS, (int)MID, (int)FFN, 0, 0, 0);
    // res2 = res1 + ffn ; norm2 = LN(res2)
    k_res_ln<<<dim3(BS), blk, 0, stream>>>(res1_f, ffn_f, g2, be2, res2_f, nrm2_b, (int)MID);
    // out = norm2 @ w_h^T + b_h  (fp32 straight to d_out)
    k_gemm_wmma<true, false><<<dim3(DOUT / 128, BS / 128, 1), blk, 0, stream>>>(
        nrm2_b, wh_b, b_h, (float*)d_out, nullptr, (int)BS, (int)DOUT, (int)MID, 0, 0, 0);
}
